// GaussianEnhancedAttention_55009941127191
// MI455X (gfx1250) — compile-verified, matrix operand order checked by reading, not run on hardware
//
#include <hip/hip_runtime.h>
#include <hip/hip_bf16.h>

// ---------------------------------------------------------------------------
// GaussianEnhancedAttention for MI455X (gfx1250, wave32, WMMA)
// B=2, N=2048, D=1024, H=16, HD=64
// Attention computes S^T = K*Q^T so softmax rows live inside a lane pair and
// the P C-layout coincides with the WMMA B-fragment layout (no LDS shuffle).
// ---------------------------------------------------------------------------

typedef __attribute__((ext_vector_type(16))) _Float16 v16h;
typedef __attribute__((ext_vector_type(8)))  _Float16 v8h;
typedef __attribute__((ext_vector_type(4)))  _Float16 v4h;
typedef __attribute__((ext_vector_type(8)))  float    v8f;

static constexpr int B_  = 2;
static constexpr int SEQ = 2048;
static constexpr int D_  = 1024;
static constexpr int H_  = 16;
static constexpr int HD_ = 64;

// ---------------------------------------------------------------------------
// WMMA helpers
// ---------------------------------------------------------------------------
__device__ inline v8f wmma_f32_f16(v16h a, v16h b, v8f c) {
  // 8 args: (neg_a, A, neg_b, B, c_mod, C, reuse_a, reuse_b)
  return __builtin_amdgcn_wmma_f32_16x16x32_f16(false, a, false, b, (short)0, c,
                                                false, false);
}

// Load a 16x32 f16 A-style fragment (also used for B via "B mirrors A" with
// n <-> m roles swapped) from a row-major matrix with leading dim `ld`.
// `base` points at element (row0, k0). ISA layout (CDNA5 ISA, 16-bit A 16x32):
//   lane L (<16):  row M=L,   K = {k0+0..7,  k0+16..23}
//   lane L (>=16): row M=L-16, K = {k0+8..15, k0+24..31}
// -> two contiguous 16-byte (v8h) loads per lane.
__device__ inline v16h load_frag16x32(const _Float16* __restrict__ base, int ld,
                                      int lane) {
  const _Float16* p = base + (size_t)(lane & 15) * (size_t)ld + ((lane >> 4) & 1) * 8;
  v8h lo = *(const v8h*)(p);
  v8h hi = *(const v8h*)(p + 16);
  v16h r;
#pragma unroll
  for (int i = 0; i < 8; ++i) { r[i] = lo[i]; r[i + 8] = hi[i]; }
  return r;
}

// ---------------------------------------------------------------------------
// f32 -> f16 convert (vectorized x4)
// ---------------------------------------------------------------------------
__global__ __launch_bounds__(256)
void cvt_f32_f16_kernel(const float* __restrict__ in, _Float16* __restrict__ out,
                        int n) {
  int i = (blockIdx.x * 256 + threadIdx.x) * 4;
  if (i + 3 < n) {
    float4 f = *(const float4*)(in + i);
    v4h h = {(_Float16)f.x, (_Float16)f.y, (_Float16)f.z, (_Float16)f.w};
    *(v4h*)(out + i) = h;
  }
}

// ---------------------------------------------------------------------------
// Transposing f32->f16 convert for DxD weights: out[n*D + k] = in[k*D + n]
// ---------------------------------------------------------------------------
__global__ __launch_bounds__(256)
void transpose_cvt_kernel(const float* __restrict__ in, _Float16* __restrict__ out) {
  __shared__ float tile[32][33];
  int x = blockIdx.x * 32 + threadIdx.x;   // input col
  int y0 = blockIdx.y * 32 + threadIdx.y;  // input row base
#pragma unroll
  for (int i = 0; i < 32; i += 8)
    tile[threadIdx.y + i][threadIdx.x] = in[(size_t)(y0 + i) * D_ + x];
  __syncthreads();
  int xo  = blockIdx.y * 32 + threadIdx.x;  // output col (= input row)
  int yo0 = blockIdx.x * 32 + threadIdx.y;  // output row (= input col)
#pragma unroll
  for (int i = 0; i < 32; i += 8)
    out[(size_t)(yo0 + i) * D_ + xo] = (_Float16)tile[threadIdx.x][threadIdx.y + i];
}

// ---------------------------------------------------------------------------
// WMMA GEMM: C[M x Nc] = A[M x K] * B[K x Nc] + bias, with B supplied as
// BT[Nc x K] row-major. Each wave: 16(M) x 64(N) tile; 8 waves/block ->
// 64(M) x 128(N) block tile.  MODE: 0 = f16 out in [b,h,n,hd] head layout
//                                   1 = f16 out in [b,h,hd,n] (transposed V)
//                                   2 = f32 out row-major [M x D]
// ---------------------------------------------------------------------------
template <int MODE>
__global__ __launch_bounds__(256)
void gemm16_kernel(const _Float16* __restrict__ A, const _Float16* __restrict__ BT,
                   const float* __restrict__ bias, void* __restrict__ Out, int K) {
  const int lane = threadIdx.x & 31;
  const int wave = threadIdx.x >> 5;
  const int wm = wave & 3;   // 0..3 (M dir)
  const int wn = wave >> 2;  // 0..1 (N dir)
  const int row0 = blockIdx.x * 64 + wm * 16;
  const int col0 = blockIdx.y * 128 + wn * 64;

  v8f acc[4] = {};
  const _Float16* Arow = A + (size_t)row0 * K;

  for (int k0 = 0; k0 < K; k0 += 32) {
    if (k0 + 64 <= K)  // speculative prefetch of next A chunk
      __builtin_prefetch(Arow + (size_t)(lane & 15) * K + k0 + 32, 0, 1);
    v16h a = load_frag16x32(Arow + k0, K, lane);
#pragma unroll
    for (int t = 0; t < 4; ++t) {
      v16h b = load_frag16x32(BT + (size_t)(col0 + t * 16) * K + k0, K, lane);
      acc[t] = wmma_f32_f16(a, b, acc[t]);
    }
  }

  const int halfsel = (lane >> 4) & 1;
  const int nl = lane & 15;
#pragma unroll
  for (int t = 0; t < 4; ++t) {
    const int n = col0 + t * 16 + nl;
    const float bv = bias[n];
#pragma unroll
    for (int v = 0; v < 8; ++v) {
      const int m = row0 + v + 8 * halfsel;
      const float val = acc[t][v] + bv;
      if constexpr (MODE == 2) {
        ((float*)Out)[(size_t)m * D_ + n] = val;
      } else {
        const int bi = m / SEQ, nn = m % SEQ;
        const int h = n / HD_, hd = n % HD_;
        size_t idx;
        if constexpr (MODE == 0)
          idx = (((size_t)bi * H_ + h) * SEQ + nn) * HD_ + hd;
        else
          idx = (((size_t)bi * H_ + h) * HD_ + hd) * SEQ + nn;
        ((_Float16*)Out)[idx] = (_Float16)val;
      }
    }
  }
}

// ---------------------------------------------------------------------------
// Flash attention with WMMA, transposed-score formulation.
// grid = (B*H, SEQ/64); block = 128 (4 waves). Each wave: 16 queries.
// Per 32-key block:
//   S^T tiles (key x query) = wmma(K frags, Q frags)      -> 4 wmma
//   softmax row = one lane pair -> in-lane reduce + 1 shfl_xor(16)
//   P (C-layout) == B-fragment layout -> direct f16 pack  -> no LDS
//   O^T (hd x query) += wmma(Vt frags, P frag)            -> 4 wmma
// ---------------------------------------------------------------------------
__global__ __launch_bounds__(128)
void attn_kernel(const _Float16* __restrict__ Qh, const _Float16* __restrict__ Kh,
                 const _Float16* __restrict__ Vt, const float* __restrict__ Bg,
                 const float* __restrict__ lamp, _Float16* __restrict__ Ah) {
  const int lane = threadIdx.x & 31;
  const int wave = threadIdx.x >> 5;
  const int bh = blockIdx.x;
  const int b = bh >> 4;   // H_ == 16
  const int h = bh & 15;
  const int q0 = blockIdx.y * 64 + wave * 16;

  const float lam = lamp[0];
  const float scale = 0.125f;  // 1/sqrt(64)

  const _Float16* Qp = Qh + (size_t)bh * SEQ * HD_;
  const _Float16* Kp = Kh + (size_t)bh * SEQ * HD_;
  const _Float16* Vp = Vt + (size_t)bh * HD_ * SEQ;

  const int halfsel = (lane >> 4) & 1;
  const int nl = lane & 15;   // query within tile (all VGPRs of this lane)

  // Gaussian-bias row for this lane's query: 8 consecutive floats per (t,half)
  const float* bgrow = Bg + ((size_t)b * SEQ + (q0 + nl)) * SEQ;

  // Q fragments (queries as B-matrix columns): 16 rows x HD(64), 2 K-chunks
  const v16h qb0 = load_frag16x32(Qp + (size_t)q0 * HD_ + 0, HD_, lane);
  const v16h qb1 = load_frag16x32(Qp + (size_t)q0 * HD_ + 32, HD_, lane);

  float mrow = -1e30f, lrow = 0.0f;  // per-lane (per-query) softmax state
  v8f o[4] = {};                     // O^T: 4 hd-chunks x 16 queries

  for (int j0 = 0; j0 < SEQ; j0 += 32) {
    // ---- S^T = K * Q^T for two 16-key tiles ------------------------------
    v8f s[2];
#pragma unroll
    for (int t = 0; t < 2; ++t) {
      v16h ka0 = load_frag16x32(Kp + (size_t)(j0 + t * 16) * HD_ + 0, HD_, lane);
      v16h ka1 = load_frag16x32(Kp + (size_t)(j0 + t * 16) * HD_ + 32, HD_, lane);
      v8f acc = {};
      acc = wmma_f32_f16(ka0, qb0, acc);
      acc = wmma_f32_f16(ka1, qb1, acc);
      s[t] = acc;
    }

    // ---- scale + Gaussian bias (contiguous per lane) ---------------------
    // s[t][v] is key j0 + 16t + 8*halfsel + v for query q0+nl.
#pragma unroll
    for (int t = 0; t < 2; ++t) {
      const float* bp = bgrow + j0 + t * 16 + 8 * halfsel;
      float4 b0 = *(const float4*)(bp);
      float4 b1 = *(const float4*)(bp + 4);
      s[t][0] = s[t][0] * scale + lam * b0.x;
      s[t][1] = s[t][1] * scale + lam * b0.y;
      s[t][2] = s[t][2] * scale + lam * b0.z;
      s[t][3] = s[t][3] * scale + lam * b0.w;
      s[t][4] = s[t][4] * scale + lam * b1.x;
      s[t][5] = s[t][5] * scale + lam * b1.y;
      s[t][6] = s[t][6] * scale + lam * b1.z;
      s[t][7] = s[t][7] * scale + lam * b1.w;
    }

    // ---- online softmax: in-lane reduce + single cross-half exchange -----
    float rmax = s[0][0];
#pragma unroll
    for (int v = 1; v < 8; ++v) rmax = fmaxf(rmax, s[0][v]);
#pragma unroll
    for (int v = 0; v < 8; ++v) rmax = fmaxf(rmax, s[1][v]);
    rmax = fmaxf(rmax, __shfl_xor(rmax, 16, 32));

    const float mnew = fmaxf(mrow, rmax);
    const float alpha = __expf(mrow - mnew);
    mrow = mnew;

    float rsum = 0.0f;
#pragma unroll
    for (int t = 0; t < 2; ++t) {
#pragma unroll
      for (int v = 0; v < 8; ++v) {
        s[t][v] = __expf(s[t][v] - mnew);
        rsum += s[t][v];
      }
    }
    rsum += __shfl_xor(rsum, 16, 32);
    lrow = lrow * alpha + rsum;

    // ---- rescale O^T by per-query alpha (per-lane scalar) ----------------
#pragma unroll
    for (int c = 0; c < 4; ++c) {
#pragma unroll
      for (int v = 0; v < 8; ++v) o[c][v] *= alpha;
    }

    // ---- pack P: C-layout of S^T coincides with the B-fragment layout ----
    // lane n:    e<8 -> keys 0..7   (tile0), e>=8 -> keys 16..23 (tile1)
    // lane n+16: e<8 -> keys 8..15  (tile0), e>=8 -> keys 24..31 (tile1)
    v16h pb;
#pragma unroll
    for (int v = 0; v < 8; ++v) {
      pb[v]     = (_Float16)s[0][v];
      pb[v + 8] = (_Float16)s[1][v];
    }

    // ---- O^T += V^T * P^T  (Vt rows contiguous over keys) ----------------
#pragma unroll
    for (int c = 0; c < 4; ++c) {
      v16h va = load_frag16x32(Vp + (size_t)(c * 16) * SEQ + j0, SEQ, lane);
      o[c] = wmma_f32_f16(va, pb, o[c]);
    }
  }

  // ---- finalize: O^T / l, contiguous v8h stores per (chunk, half) --------
  const float linv = 1.0f / lrow;
  _Float16* outrow = Ah + ((size_t)(b * SEQ + q0 + nl)) * D_ + h * HD_;
#pragma unroll
  for (int c = 0; c < 4; ++c) {
    v8h hv;
#pragma unroll
    for (int v = 0; v < 8; ++v) hv[v] = (_Float16)(o[c][v] * linv);
    *(v8h*)(outrow + c * 16 + 8 * halfsel) = hv;
  }
}

// ---------------------------------------------------------------------------
// Host-side launch
// ---------------------------------------------------------------------------
extern "C" void kernel_launch(void* const* d_in, const int* in_sizes, int n_in,
                              void* d_out, int out_size, void* d_ws, size_t ws_size,
                              hipStream_t stream) {
  (void)in_sizes; (void)n_in; (void)out_size; (void)ws_size;

  const float* x   = (const float*)d_in[0];
  const float* Bg  = (const float*)d_in[1];
  const float* Wq  = (const float*)d_in[2];
  const float* bq  = (const float*)d_in[3];
  const float* Wk  = (const float*)d_in[4];
  const float* bk  = (const float*)d_in[5];
  const float* Wv  = (const float*)d_in[6];
  const float* bv  = (const float*)d_in[7];
  const float* Wo  = (const float*)d_in[8];
  const float* bo  = (const float*)d_in[9];
  const float* lam = (const float*)d_in[10];
  float* out = (float*)d_out;

  // workspace carve-up (f16): 48 MB total
  const size_t XH = (size_t)B_ * SEQ * D_;  // 4M elements
  const size_t WT = (size_t)D_ * D_;        // 1M elements
  _Float16* xh  = (_Float16*)d_ws;
  _Float16* wqT = xh + XH;
  _Float16* wkT = wqT + WT;
  _Float16* wvT = wkT + WT;
  _Float16* woT = wvT + WT;
  _Float16* Qh  = woT + WT;
  _Float16* Kh  = Qh + XH;
  _Float16* Vt  = Kh + XH;
  _Float16* Ah  = Vt + XH;

  // 1) convert activations, transpose+convert weights
  cvt_f32_f16_kernel<<<(unsigned)(XH / 1024), 256, 0, stream>>>(x, xh, (int)XH);
  dim3 tb(32, 8), tg(D_ / 32, D_ / 32);
  transpose_cvt_kernel<<<tg, tb, 0, stream>>>(Wq, wqT);
  transpose_cvt_kernel<<<tg, tb, 0, stream>>>(Wk, wkT);
  transpose_cvt_kernel<<<tg, tb, 0, stream>>>(Wv, wvT);
  transpose_cvt_kernel<<<tg, tb, 0, stream>>>(Wo, woT);

  // 2) Q/K/V projections (WMMA)
  dim3 gb(256), gg((B_ * SEQ) / 64, D_ / 128);
  gemm16_kernel<0><<<gg, gb, 0, stream>>>(xh, wqT, bq, (void*)Qh, D_);
  gemm16_kernel<0><<<gg, gb, 0, stream>>>(xh, wkT, bk, (void*)Kh, D_);
  gemm16_kernel<1><<<gg, gb, 0, stream>>>(xh, wvT, bv, (void*)Vt, D_);

  // 3) flash attention with Gaussian bias (WMMA, transposed scores)
  attn_kernel<<<dim3(B_ * H_, SEQ / 64), 128, 0, stream>>>(Qh, Kh, Vt, Bg, lam, Ah);

  // 4) output projection, f32 result (WMMA)
  gemm16_kernel<2><<<gg, gb, 0, stream>>>(Ah, woT, bo, (void*)out, D_);
}
